// CFRMPhilosophyClassifier_81123342287194
// MI455X (gfx1250) — matrix operand order repaired and checked by reading, not compiled, not run on hardware
//
#include <hip/hip_runtime.h>

// ---------------- problem constants ----------------
constexpr int B  = 128;   // batch
constexpr int S  = 512;   // sequence
constexpr int H  = 128;   // hidden
constexpr int C  = 16;    // clusters
constexpr int CI = 260;   // 2H+4 controller input
constexpr int NC = 8;     // classes
constexpr int BT = 16;    // batches per workgroup (WMMA M)
constexpr int KP = 288;   // ci K padded to 9 x 32
constexpr int NH = 208;   // packed heads width (13 x 16)
constexpr float EPS = 1e-4f;

typedef __attribute__((ext_vector_type(16))) __bf16 v16bf;
typedef __attribute__((ext_vector_type(8)))  float  v8f;

#define DEVFN __device__ __forceinline__

// ---------------- WMMA helpers (gfx1250, wave32) ----------------
DEVFN v8f wmma_bf16(v16bf a, v16bf b, v8f c) {
  // D = A(16x32 bf16) x B(32x16 bf16) + C(16x16 f32)
  return __builtin_amdgcn_wmma_f32_16x16x32_bf16(false, a, false, b, (short)0, c, false, false);
}

// A fragment, 16x32 bf16, row-major source with stride lda.
// Layout: lanes0-15 rows, element e of lane half h holds K=16h+e (contiguous, matches B loaders).
DEVFN v16bf load_a_bf(const __bf16* A, int lda, int kbase, int lane) {
  int row = lane & 15, half = lane >> 4;
  const __bf16* Ar = A + row * lda + kbase + half * 16;
  v16bf a;
#pragma unroll
  for (int e = 0; e < 16; ++e) {
    int v = e >> 1;
    int k = e;
    a[e] = Ar[k];
  }
  return a;
}

// A fragment from f32 LDS with K zero-padding beyond klim (converted bf16).
DEVFN v16bf load_a_f32(const float* A, int lda, int kbase, int klim, int lane) {
  int row = lane & 15, half = lane >> 4;
  v16bf a;
#pragma unroll
  for (int e = 0; e < 16; ++e) {
    int v = e >> 1;
    int k = kbase + half * 16 + e;
    a[e] = (k < klim) ? (__bf16)A[row * lda + k] : (__bf16)0.0f;
  }
  return a;
}

// B fragment, 32x16, from PRE-TRANSPOSED bf16 weights Wt[N][K] (ldk = K stride).
// Per-lane data is 16 consecutive bf16 -> one aligned 32B load.
DEVFN v16bf load_bt_bf(const __bf16* Wt, int ldk, int kbase, int n0, int lane) {
  int col = n0 + (lane & 15), half = lane >> 4;
  const __bf16* p = Wt + (size_t)col * ldk + kbase + half * 16;
  return *(const v16bf*)p;
}

// B fragment, 32x16, from f32 row-major [K][N] with zero pad beyond klim.
DEVFN v16bf load_b_f32(const float* Bm, int ldb, int kbase, int klim, int n0, int lane) {
  int col = n0 + (lane & 15), half = lane >> 4;
  v16bf b;
#pragma unroll
  for (int e = 0; e < 16; ++e) {
    int k = kbase + half * 16 + e;
    b[e] = (k < klim) ? (__bf16)Bm[k * ldb + col] : (__bf16)0.0f;
  }
  return b;
}

// B fragment, 32x16, where B[k][n] = M[n][k] (transposed read of f32 [N][K]).
DEVFN v16bf load_bt_f32(const float* M, int ldm, int kbase, int n0, int lane) {
  int col = n0 + (lane & 15), half = lane >> 4;
  v16bf b;
#pragma unroll
  for (int e = 0; e < 16; ++e) {
    int k = kbase + half * 16 + e;
    b[e] = (__bf16)M[col * ldm + k];
  }
  return b;
}

DEVFN float sigmoidf_(float x) { return 1.f / (1.f + __expf(-x)); }

// ---------------- scan-kernel shared memory ----------------
struct Smem {
  float centers[BT][C][H];           // 128 KB, recurrent state
  float core[BT][H];                 // 8 KB
  float heads[BT][NH];               // 13 KB
  float compat[BT][C][C];            // 16 KB (compat -> mixing in place)
  float spreads[BT][C], masses[BT][C], alpha[BT][C], scores[BT][C];
  float sqd[BT][C], sq[BT][C], msv[BT][C], mmv[BT][C];
  float valid[BT], nov[BT], relax[BT], en[BT];
  __bf16 ci[BT][KP];                 // WMMA A operand, K padded with zeros
  __bf16 ctrl1[BT][H];
  __bf16 ctrl[BT][H];
};

// ---------------- kernel 0: weight prep (bf16, transposed [N][K]) ----------------
constexpr int N_W1P = 128 * 288;     // W1^T padded K
constexpr int N_W2  = 128 * 128;
constexpr int N_WH  = NH * 128;      // packed heads^T
constexpr int N_BH  = NH;
constexpr int N_WCC = 2048 * 128;
constexpr int N_PREP = N_W1P + N_W2 + N_WH + N_BH + N_WCC;

__global__ void prep_kernel(const float* ctrl_w1, const float* ctrl_w2,
                            const float* gate_w, const float* gate_b,
                            const float* assign_w, const float* assign_b,
                            const float* nov_w, const float* nov_b,
                            const float* relax_w, const float* relax_b,
                            const float* cs_w, const float* cs_b,
                            const float* md_w, const float* md_b,
                            const float* att_w, const float* att_b,
                            const float* cc_w,
                            __bf16* W1p, __bf16* W2t, __bf16* Wht, float* bh, __bf16* Wcct) {
  for (int idx = blockIdx.x * blockDim.x + threadIdx.x; idx < N_PREP;
       idx += gridDim.x * blockDim.x) {
    int i = idx;
    if (i < N_W1P) { int n = i / 288, k = i % 288;
      W1p[i] = (__bf16)((k < CI) ? ctrl_w1[k * H + n] : 0.f); continue; }
    i -= N_W1P;
    if (i < N_W2) { int n = i / 128, k = i % 128;
      W2t[i] = (__bf16)ctrl_w2[k * H + n]; continue; }
    i -= N_W2;
    if (i < N_WH) { int c2 = i / 128, hh = i % 128; float v = 0.f;
      if      (c2 < 16)              v = gate_w[hh * 16 + c2];
      else if (c2 < 32)              v = assign_w[hh * 16 + c2 - 16];
      else if (c2 == 32)             v = nov_w[hh];
      else if (c2 == 33)             v = relax_w[hh];
      else if (c2 >= 48 && c2 < 64)  v = cs_w[hh * 16 + c2 - 48];
      else if (c2 >= 64 && c2 < 80)  v = md_w[hh * 16 + c2 - 64];
      else if (c2 >= 80)             v = att_w[hh * 128 + c2 - 80];
      Wht[i] = (__bf16)v; continue; }
    i -= N_WH;
    if (i < N_BH) { int c2 = i; float v = 0.f;
      if      (c2 < 16)              v = gate_b[c2];
      else if (c2 < 32)              v = assign_b[c2 - 16];
      else if (c2 == 32)             v = nov_b[0];
      else if (c2 == 33)             v = relax_b[0];
      else if (c2 >= 48 && c2 < 64)  v = cs_b[c2 - 48];
      else if (c2 >= 64 && c2 < 80)  v = md_b[c2 - 64];
      else if (c2 >= 80)             v = att_b[c2 - 80];
      bh[i] = v; continue; }
    i -= N_BH;
    { int n = i / 128, k = i % 128; Wcct[i] = (__bf16)cc_w[k * 2048 + n]; }
  }
}

// ---------------- kernel 1: embedding + LayerNorm -> bf16 te ----------------
__global__ __launch_bounds__(128) void embed_ln_kernel(const int* tokens, const float* emb,
                                                       const float* ln_g, const float* ln_b,
                                                       __bf16* te) {
  __shared__ float red[128];
  int bs = blockIdx.x, h = threadIdx.x;
  int tok = tokens[bs];
  float e = (tok == 0) ? 0.f : emb[(size_t)tok * H + h];   // emb row 0 forced to zero
  red[h] = e; __syncthreads();
  for (int s2 = 64; s2 > 0; s2 >>= 1) { if (h < s2) red[h] += red[h + s2]; __syncthreads(); }
  float mu = red[0] * (1.f / H); __syncthreads();
  float dl = e - mu; red[h] = dl * dl; __syncthreads();
  for (int s2 = 64; s2 > 0; s2 >>= 1) { if (h < s2) red[h] += red[h + s2]; __syncthreads(); }
  float var = red[0] * (1.f / H);
  float v = dl * rsqrtf(var + 1e-5f) * ln_g[h] + ln_b[h];
  te[(size_t)bs * H + h] = (__bf16)v;
}

// ---------------- kernel 2: the recurrent scan (8 blocks x 16 batches) ----------------
__global__ __launch_bounds__(256) void scan_kernel(const int* tokens, const __bf16* te,
                                                   const __bf16* W1p, const float* b1,
                                                   const __bf16* W2t, const float* b2,
                                                   const __bf16* Wht, const float* bh,
                                                   const __bf16* Wcct, const float* ccb,
                                                   float* cfin, float* sfin, float* mfin) {
  extern __shared__ char smem_raw[];
  Smem& s = *reinterpret_cast<Smem*>(smem_raw);
  const int tid = threadIdx.x, lane = tid & 31, wave = tid >> 5;
  const int b0 = blockIdx.x * BT;
  const int bq = tid >> 4, cq = tid & 15;      // (b,c) pair per thread, 256 = 16x16

  for (int i = tid; i < BT * C * H; i += 256) (&s.centers[0][0][0])[i] = 0.f;
  for (int i = tid; i < BT * C; i += 256) { (&s.spreads[0][0])[i] = 1.f; (&s.masses[0][0])[i] = 0.f; }
  for (int i = tid; i < BT * KP; i += 256) (&s.ci[0][0])[i] = (__bf16)0.f;  // pad cols stay zero
  __syncthreads();

  for (int t = 0; t < S; ++t) {
    // ---- summarize: scores ----
    { float sp = s.spreads[bq][cq];
      s.scores[bq][cq] = s.masses[bq][cq] + __logf(1.f / (sp + EPS) + EPS); }
    __syncthreads();
    if (tid < BT) {                         // alpha softmax + energy (logsumexp)
      int b = tid; float mx = -1e30f, m2 = -1e30f;
      for (int c = 0; c < C; ++c) { mx = fmaxf(mx, s.scores[b][c]); m2 = fmaxf(m2, s.masses[b][c]); }
      float sm = 0.f, s2 = 0.f;
      for (int c = 0; c < C; ++c) { sm += __expf(s.scores[b][c] - mx); s2 += __expf(s.masses[b][c] - m2); }
      float inv = 1.f / sm;
      for (int c = 0; c < C; ++c) s.alpha[b][c] = __expf(s.scores[b][c] - mx) * inv;
      s.en[b] = m2 + __logf(s2);
      s.valid[b] = (tokens[(b0 + b) * S + t] != 0) ? 1.f : 0.f;
    }
    __syncthreads();
    // ---- core + ci (token part + core part) ----
    for (int o = tid; o < BT * H; o += 256) {
      int b = o >> 7, h = o & 127;
      float acc = 0.f;
      for (int c = 0; c < C; ++c) acc += s.alpha[b][c] * s.centers[b][c][h];
      s.core[b][h] = acc;
      s.ci[b][H + h] = (__bf16)acc;
      s.ci[b][h] = te[((size_t)(b0 + b) * S + t) * H + h];
    }
    __syncthreads();
    // ---- per-cluster mean-square distance to core ----
    { float acc = 0.f;
      for (int h = 0; h < H; ++h) { float d = s.centers[bq][cq][h] - s.core[bq][h]; acc += d * d; }
      s.sqd[bq][cq] = acc * (1.f / H); }
    __syncthreads();
    if (tid < BT) {                         // u, d, entropy -> ci tail
      int b = tid; float u = 0.f, d = 0.f, ent = 0.f;
      for (int c = 0; c < C; ++c) {
        float a = s.alpha[b][c];
        u += a * s.spreads[b][c]; d += a * s.sqd[b][c];
        ent -= a * __logf(fmaxf(a, 1e-8f));
      }
      s.ci[b][2 * H + 0] = (__bf16)u;  s.ci[b][2 * H + 1] = (__bf16)d;
      s.ci[b][2 * H + 2] = (__bf16)s.en[b]; s.ci[b][2 * H + 3] = (__bf16)ent;
    }
    __syncthreads();
    // ---- GEMM1: ctrl1 = tanh(ci @ W1 + b1), 16x288x128 ----
    { int n0 = wave * 16; v8f acc = {};
#pragma unroll
      for (int kt = 0; kt < KP; kt += 32)
        acc = wmma_bf16(load_a_bf(&s.ci[0][0], KP, kt, lane),
                        load_bt_bf(W1p, KP, kt, n0, lane), acc);
      int col = n0 + (lane & 15), half = lane >> 4; float bias = b1[col];
#pragma unroll
      for (int r = 0; r < 8; ++r) s.ctrl1[r + half * 8][col] = (__bf16)tanhf(acc[r] + bias);
    }
    __syncthreads();
    // ---- GEMM2: ctrl = tanh(ctrl1 @ W2 + b2), 16x128x128 ----
    { int n0 = wave * 16; v8f acc = {};
#pragma unroll
      for (int kt = 0; kt < H; kt += 32)
        acc = wmma_bf16(load_a_bf(&s.ctrl1[0][0], H, kt, lane),
                        load_bt_bf(W2t, H, kt, n0, lane), acc);
      int col = n0 + (lane & 15), half = lane >> 4; float bias = b2[col];
#pragma unroll
      for (int r = 0; r < 8; ++r) s.ctrl[r + half * 8][col] = (__bf16)tanhf(acc[r] + bias);
    }
    __syncthreads();
    // ---- GEMM3: packed heads = ctrl @ Wh + bh, 16x128x208 ----
    for (int nt = wave; nt < 13; nt += 8) {
      int n0 = nt * 16; v8f acc = {};
#pragma unroll
      for (int kt = 0; kt < H; kt += 32)
        acc = wmma_bf16(load_a_bf(&s.ctrl[0][0], H, kt, lane),
                        load_bt_bf(Wht, H, kt, n0, lane), acc);
      int col = n0 + (lane & 15), half = lane >> 4; float bias = bh[col];
#pragma unroll
      for (int r = 0; r < 8; ++r) s.heads[r + half * 8][col] = acc[r] + bias;
    }
    __syncthreads();
    // ---- head activations + spreads/masses ss-update ----
    if (tid < BT) {
      int b = tid; float valid = s.valid[b];
      float mx = -1e30f;
      for (int c = 0; c < C; ++c) mx = fmaxf(mx, s.heads[b][16 + c]);
      float ex[C]; float sm = 0.f;
      for (int c = 0; c < C; ++c) { ex[c] = __expf(s.heads[b][16 + c] - mx); sm += ex[c]; }
      float inv = 1.f / sm;
      s.nov[b]   = sigmoidf_(s.heads[b][32]) * valid;
      s.relax[b] = sigmoidf_(s.heads[b][33]) * valid;
      for (int c = 0; c < C; ++c) {
        float gate = sigmoidf_(s.heads[b][c]) * valid;
        float ss = gate * ex[c] * inv;
        s.alpha[b][c] = ss;   // reuse alpha buffer as ss (alpha no longer needed this step)
        float x = s.heads[b][48 + c];
        float cand_s = ((x > 20.f) ? x : __logf(1.f + __expf(x))) + EPS;  // softplus+EPS
        float mdelta = tanhf(s.heads[b][64 + c]);
        s.spreads[b][c] += ss * (cand_s - s.spreads[b][c]);
        s.masses[b][c]  += ss * mdelta;
      }
    }
    __syncthreads();
    // ---- GEMM4: cand_c = ctrl @ cc_w + cc_b (16x128x2048) fused center update ----
    for (int nt = wave; nt < 128; nt += 8) {
      int n0 = nt * 16; v8f acc = {};
#pragma unroll
      for (int kt = 0; kt < H; kt += 32)
        acc = wmma_bf16(load_a_bf(&s.ctrl[0][0], H, kt, lane),
                        load_bt_bf(Wcct, H, kt, n0, lane), acc);
      int col = n0 + (lane & 15), half = lane >> 4;
      int cc = col >> 7, hh = col & 127; float bias = ccb[col];
#pragma unroll
      for (int r = 0; r < 8; ++r) {
        int m = r + half * 8;
        float cen = s.centers[m][cc][hh];
        s.centers[m][cc][hh] = cen + s.alpha[m][cc] * ((acc[r] + bias) - cen);  // ss stored in alpha
      }
    }
    __syncthreads();
    // ---- attractor pull ----
    for (int o = tid; o < BT * C * H; o += 256) {
      int b = o >> 11, rem = o & 2047, c = rem >> 7, h = rem & 127;
      float cen = s.centers[b][c][h];
      s.centers[b][c][h] = cen + 0.1f * s.nov[b] * (s.heads[b][80 + h] - cen);
    }
    __syncthreads();
    // ---- interaction: squared norms ----
    { float acc = 0.f;
      for (int h = 0; h < H; ++h) { float v = s.centers[bq][cq][h]; acc += v * v; }
      s.sq[bq][cq] = acc; }
    __syncthreads();
    // ---- Gram = centers x centers^T via WMMA -> compat ----
    for (int bb = wave; bb < BT; bb += 8) {
      v8f acc = {};
#pragma unroll
      for (int kt = 0; kt < H; kt += 32)
        acc = wmma_bf16(load_a_f32(&s.centers[bb][0][0], H, kt, H, lane),
                        load_bt_f32(&s.centers[bb][0][0], H, kt, 0, lane), acc);
      int j = lane & 15, half = lane >> 4;
#pragma unroll
      for (int r = 0; r < 8; ++r) {
        int i = r + half * 8;
        float d2 = fmaxf(s.sq[bb][i] + s.sq[bb][j] - 2.f * acc[r], 0.f);
        float scale = s.spreads[bb][i] + s.spreads[bb][j] + EPS;
        s.compat[bb][i][j] = -d2 / scale + s.masses[bb][j];
      }
    }
    __syncthreads();
    // ---- mixing = softmax(compat) rows, in place ----
    { int b = bq, i = cq;
      float mx = -1e30f;
      for (int j = 0; j < C; ++j) mx = fmaxf(mx, s.compat[b][i][j]);
      float sm = 0.f;
      for (int j = 0; j < C; ++j) sm += __expf(s.compat[b][i][j] - mx);
      float inv = 1.f / sm;
      for (int j = 0; j < C; ++j) s.compat[b][i][j] = __expf(s.compat[b][i][j] - mx) * inv; }
    __syncthreads();
    // ---- mixed spreads/masses ----
    { float ms = 0.f, mm = 0.f;
      for (int j = 0; j < C; ++j) { float w = s.compat[bq][cq][j]; ms += w * s.spreads[bq][j]; mm += w * s.masses[bq][j]; }
      s.msv[bq][cq] = ms; s.mmv[bq][cq] = mm; }
    // ---- mixed centers via WMMA (mixing 16x16 @ centers 16x128), blend in place ----
    // wave w owns column tile n0 = 16*w for every batch -> disjoint writes, no barrier needed
    for (int task = wave; task < BT * 8; task += 8) {
      int bb = task >> 3, n0 = (task & 7) * 16;
      v8f acc = wmma_bf16(load_a_f32(&s.compat[bb][0][0], C, 0, C, lane),
                          load_b_f32(&s.centers[bb][0][0], H, 0, C, n0, lane), (v8f){});
      int h = n0 + (lane & 15), half = lane >> 4;
      float rel = s.relax[bb];
#pragma unroll
      for (int r = 0; r < 8; ++r) {
        int i = r + half * 8;
        s.centers[bb][i][h] = (1.f - rel) * s.centers[bb][i][h] + rel * acc[r];
      }
    }
    __syncthreads();
    // ---- blend spreads/masses, rebuild scores ----
    { float rel = s.relax[bq];
      float sp = (1.f - rel) * s.spreads[bq][cq] + rel * s.msv[bq][cq];
      float ma = (1.f - rel) * s.masses[bq][cq] + rel * s.mmv[bq][cq];
      s.spreads[bq][cq] = sp; s.masses[bq][cq] = ma;
      s.scores[bq][cq] = ma + __logf(1.f / (sp + EPS) + EPS); }
    __syncthreads();
    if (tid < BT) {                          // ca softmax + spread shrink
      int b = tid; float mx = -1e30f;
      for (int c = 0; c < C; ++c) mx = fmaxf(mx, s.scores[b][c]);
      float sm = 0.f;
      for (int c = 0; c < C; ++c) sm += __expf(s.scores[b][c] - mx);
      float inv = 1.f / sm, valid = s.valid[b];
      for (int c = 0; c < C; ++c) {
        float ca = __expf(s.scores[b][c] - mx) * inv;
        s.spreads[b][c] = s.spreads[b][c] * (1.f - 0.05f * ca * valid) + EPS;
      }
    }
    __syncthreads();
  }

  // ---- write final state ----
  for (int o = tid; o < BT * C * H; o += 256) cfin[(size_t)b0 * C * H + o] = (&s.centers[0][0][0])[o];
  for (int o = tid; o < BT * C; o += 256) { sfin[b0 * C + o] = (&s.spreads[0][0])[o];
                                            mfin[b0 * C + o] = (&s.masses[0][0])[o]; }
}

// ---------------- kernel 3: final summarize + classifier head ----------------
__global__ __launch_bounds__(128) void cls_kernel(const float* cfin, const float* sfin,
                                                  const float* mfin, const float* w1,
                                                  const float* bb1, const float* w2,
                                                  const float* bb2, float* out) {
  __shared__ float cen[C][H], core[H], feats[CI], h1v[H];
  __shared__ float alpha[C], sqd[C], spv[C], mav[C];
  __shared__ float een; __shared__ int sidx;
  int b = blockIdx.x, h = threadIdx.x;
  for (int o = h; o < C * H; o += 128) (&cen[0][0])[o] = cfin[(size_t)b * C * H + o];
  if (h < C) { spv[h] = sfin[b * C + h]; mav[h] = mfin[b * C + h]; }
  __syncthreads();
  if (h == 0) {
    float sc[C]; float mx = -1e30f, m2 = -1e30f;
    for (int c = 0; c < C; ++c) { sc[c] = mav[c] + logf(1.f / (spv[c] + EPS) + EPS);
                                  mx = fmaxf(mx, sc[c]); m2 = fmaxf(m2, mav[c]); }
    float sm = 0.f, s2 = 0.f;
    for (int c = 0; c < C; ++c) { sc[c] = expf(sc[c] - mx); sm += sc[c]; s2 += expf(mav[c] - m2); }
    int bi = 0; float bv = -1.f;
    for (int c = 0; c < C; ++c) { float a = sc[c] / sm; alpha[c] = a; if (a > bv) { bv = a; bi = c; } }
    sidx = bi; een = m2 + logf(s2);
  }
  __syncthreads();
  { float acc = 0.f;
    for (int c = 0; c < C; ++c) acc += alpha[c] * cen[c][h];
    core[h] = acc; feats[h] = acc; }
  __syncthreads();
  feats[H + h] = cen[sidx][h];
  if (h < C) { float acc = 0.f;
    for (int k = 0; k < H; ++k) { float d = cen[h][k] - core[k]; acc += d * d; }
    sqd[h] = acc * (1.f / H); }
  __syncthreads();
  if (h == 0) {
    float u = 0.f, d = 0.f, ent = 0.f;
    for (int c = 0; c < C; ++c) { float a = alpha[c]; u += a * spv[c]; d += a * sqd[c];
                                  ent -= a * logf(fmaxf(a, 1e-8f)); }
    feats[256] = u; feats[257] = d; feats[258] = een; feats[259] = ent;
  }
  __syncthreads();
  { float acc = bb1[h];
    for (int k = 0; k < CI; ++k) acc += feats[k] * w1[k * H + h];
    h1v[h] = 0.5f * acc * (1.f + erff(acc * 0.70710678f)); }    // exact GELU
  __syncthreads();
  if (h < NC) {
    float acc = bb2[h];
    for (int k = 0; k < H; ++k) acc += h1v[k] * w2[k * NC + h];
    out[b * NC + h] = acc;
  }
}

// ---------------- host launcher ----------------
extern "C" void kernel_launch(void* const* d_in, const int* in_sizes, int n_in,
                              void* d_out, int out_size, void* d_ws, size_t ws_size,
                              hipStream_t stream) {
  (void)in_sizes; (void)n_in; (void)out_size; (void)ws_size;
  const int*   tokens   = (const int*)  d_in[0];
  const float* emb      = (const float*)d_in[1];
  const float* ln_g     = (const float*)d_in[2];
  const float* ln_b     = (const float*)d_in[3];
  const float* ctrl_w1  = (const float*)d_in[4];
  const float* ctrl_b1  = (const float*)d_in[5];
  const float* ctrl_w2  = (const float*)d_in[6];
  const float* ctrl_b2  = (const float*)d_in[7];
  const float* gate_w   = (const float*)d_in[8];
  const float* gate_b   = (const float*)d_in[9];
  const float* assign_w = (const float*)d_in[10];
  const float* assign_b = (const float*)d_in[11];
  const float* nov_w    = (const float*)d_in[12];
  const float* nov_b    = (const float*)d_in[13];
  const float* relax_w  = (const float*)d_in[14];
  const float* relax_b  = (const float*)d_in[15];
  const float* cc_w     = (const float*)d_in[16];
  const float* cc_b     = (const float*)d_in[17];
  const float* cs_w     = (const float*)d_in[18];
  const float* cs_b     = (const float*)d_in[19];
  const float* md_w     = (const float*)d_in[20];
  const float* md_b     = (const float*)d_in[21];
  const float* att_w    = (const float*)d_in[22];
  const float* att_b    = (const float*)d_in[23];
  const float* cls_w1   = (const float*)d_in[24];
  const float* cls_b1   = (const float*)d_in[25];
  const float* cls_w2   = (const float*)d_in[26];
  const float* cls_b2   = (const float*)d_in[27];
  float* out = (float*)d_out;

  size_t off = 0;
  auto take = [&](size_t bytes) {
    void* p = (char*)d_ws + off;
    off += (bytes + 255) & ~(size_t)255;
    return p;
  };
  __bf16* te   = (__bf16*)take((size_t)B * S * H * sizeof(__bf16));
  __bf16* W1p  = (__bf16*)take((size_t)N_W1P * sizeof(__bf16));
  __bf16* W2t  = (__bf16*)take((size_t)N_W2  * sizeof(__bf16));
  __bf16* Wht  = (__bf16*)take((size_t)N_WH  * sizeof(__bf16));
  float*  bh   = (float*) take((size_t)N_BH  * sizeof(float));
  __bf16* Wcct = (__bf16*)take((size_t)N_WCC * sizeof(__bf16));
  float*  cfin = (float*) take((size_t)B * C * H * sizeof(float));
  float*  sfin = (float*) take((size_t)B * C * sizeof(float));
  float*  mfin = (float*) take((size_t)B * C * sizeof(float));

  prep_kernel<<<256, 256, 0, stream>>>(ctrl_w1, ctrl_w2, gate_w, gate_b, assign_w, assign_b,
                                       nov_w, nov_b, relax_w, relax_b, cs_w, cs_b, md_w, md_b,
                                       att_w, att_b, cc_w, W1p, W2t, Wht, bh, Wcct);
  embed_ln_kernel<<<B * S, 128, 0, stream>>>(tokens, emb, ln_g, ln_b, te);

  (void)hipFuncSetAttribute(reinterpret_cast<const void*>(scan_kernel),
                            hipFuncAttributeMaxDynamicSharedMemorySize, (int)sizeof(Smem));
  scan_kernel<<<B / BT, 256, sizeof(Smem), stream>>>(tokens, te, W1p, ctrl_b1, W2t, ctrl_b2,
                                                     Wht, bh, Wcct, cc_b, cfin, sfin, mfin);
  cls_kernel<<<B, 128, 0, stream>>>(cfin, sfin, mfin, cls_w1, cls_b1, cls_w2, cls_b2, out);
}
